// CustomLoss_38165079392610
// MI455X (gfx1250) — compile-verified, matrix operand order checked by reading, not run on hardware
//
#include <hip/hip_runtime.h>
#include <hip/hip_bf16.h>
#include <math.h>

// Problem constants (from the reference)
#define N_E  4096
#define N_W  8192
#define KDIM 128
#define G_   1024
#define M_   64
#define J_   256

typedef __attribute__((ext_vector_type(16))) _Float16 v16h;
typedef __attribute__((ext_vector_type(8)))  float    v8f;

// ---------------- wave32 / block reductions (deterministic, fixed order) ----

__device__ __forceinline__ float wave_reduce(float v) {
#pragma unroll
    for (int off = 16; off > 0; off >>= 1) v += __shfl_xor(v, off, 32);
    return v;
}

// Result valid on threadIdx.x == 0 only. Caller must __syncthreads() before
// reusing `lds`.
__device__ __forceinline__ float block_reduce(float v, float* lds) {
    const int lane = threadIdx.x & 31;
    const int wv   = threadIdx.x >> 5;
    v = wave_reduce(v);
    if (lane == 0) lds[wv] = v;
    __syncthreads();
    float r = 0.0f;
    if (threadIdx.x == 0) {
        const int nw = blockDim.x >> 5;
        for (int i = 0; i < nw; ++i) r += lds[i];
    }
    return r;
}

// ---------------- WMMA-based streaming sum-of-squares reductions ------------
//
// Trick: A = all-ones f16 (layout-independent: every slot is 1.0), B holds 512
// squared values per wave-tile. C[i,j] += sum_k B[k,j] (f32 accumulation), so
// sum(all C entries) == 16 * sum(all B elements) regardless of the physical
// VGPR <-> matrix-element mapping. One v_wmma per 512 streamed floats.

__global__ void sumsq_diff_wmma(const float* __restrict__ a,
                                const float* __restrict__ p,
                                int ntiles, float* __restrict__ partials) {
    const int lane   = threadIdx.x & 31;
    const int wvblk  = threadIdx.x >> 5;
    const int gwave  = (blockIdx.x * blockDim.x + threadIdx.x) >> 5;
    const int nwaves = (gridDim.x * blockDim.x) >> 5;

    v16h ones;
#pragma unroll
    for (int i = 0; i < 16; ++i) ones[i] = (_Float16)1.0f;

    v8f c = {};
    for (int t = gwave; t < ntiles; t += nwaves) {
        const long long base = (long long)t * 512 + (long long)lane * 16;
        const float4* A4 = (const float4*)(a + base);
        const float4* P4 = (const float4*)(p + base);
        v16h b;
#pragma unroll
        for (int q = 0; q < 4; ++q) {
            float4 x = A4[q];
            float4 y = P4[q];
            float d0 = x.x - y.x, d1 = x.y - y.y;
            float d2 = x.z - y.z, d3 = x.w - y.w;
            b[4 * q + 0] = (_Float16)(d0 * d0);
            b[4 * q + 1] = (_Float16)(d1 * d1);
            b[4 * q + 2] = (_Float16)(d2 * d2);
            b[4 * q + 3] = (_Float16)(d3 * d3);
        }
        c = __builtin_amdgcn_wmma_f32_16x16x32_f16(false, ones, false, b,
                                                   (short)0, c, false, false);
    }

    float s = 0.0f;
#pragma unroll
    for (int i = 0; i < 8; ++i) s += c[i];
    s = wave_reduce(s);

    __shared__ float lds[8];
    if (lane == 0) lds[wvblk] = s;
    __syncthreads();
    if (threadIdx.x == 0) {
        float r = 0.0f;
        const int nw = blockDim.x >> 5;
        for (int i = 0; i < nw; ++i) r += lds[i];
        partials[blockIdx.x] = r * (1.0f / 16.0f);  // each value counted 16x in C
    }
}

__global__ void sumsq_relu_wmma(const float* __restrict__ x,
                                int ntiles, float* __restrict__ partials) {
    const int lane   = threadIdx.x & 31;
    const int wvblk  = threadIdx.x >> 5;
    const int gwave  = (blockIdx.x * blockDim.x + threadIdx.x) >> 5;
    const int nwaves = (gridDim.x * blockDim.x) >> 5;

    v16h ones;
#pragma unroll
    for (int i = 0; i < 16; ++i) ones[i] = (_Float16)1.0f;

    v8f c = {};
    for (int t = gwave; t < ntiles; t += nwaves) {
        const long long base = (long long)t * 512 + (long long)lane * 16;
        const float4* X4 = (const float4*)(x + base);
        v16h b;
#pragma unroll
        for (int q = 0; q < 4; ++q) {
            float4 v = X4[q];
            float r0 = fmaxf(v.x, 0.0f), r1 = fmaxf(v.y, 0.0f);
            float r2 = fmaxf(v.z, 0.0f), r3 = fmaxf(v.w, 0.0f);
            b[4 * q + 0] = (_Float16)(r0 * r0);
            b[4 * q + 1] = (_Float16)(r1 * r1);
            b[4 * q + 2] = (_Float16)(r2 * r2);
            b[4 * q + 3] = (_Float16)(r3 * r3);
        }
        c = __builtin_amdgcn_wmma_f32_16x16x32_f16(false, ones, false, b,
                                                   (short)0, c, false, false);
    }

    float s = 0.0f;
#pragma unroll
    for (int i = 0; i < 8; ++i) s += c[i];
    s = wave_reduce(s);

    __shared__ float lds[8];
    if (lane == 0) lds[wvblk] = s;
    __syncthreads();
    if (threadIdx.x == 0) {
        float r = 0.0f;
        const int nw = blockDim.x >> 5;
        for (int i = 0; i < nw; ++i) r += lds[i];
        partials[blockIdx.x] = r * (1.0f / 16.0f);
    }
}

// ---------------- diff_w: one workgroup per group g (W is L2-resident) ------

__global__ void diffw_kernel(const float* __restrict__ W,
                             const float* __restrict__ Sw,
                             const int* __restrict__ word_i,
                             const int* __restrict__ sample_j,
                             float* __restrict__ gterm) {
    const int g = blockIdx.x;
    const int j = sample_j[g];

    __shared__ float yj[KDIM];
    __shared__ int   wi[M_];
    __shared__ float lds[8];

    for (int k = threadIdx.x; k < KDIM; k += blockDim.x)
        yj[k] = W[(size_t)k * N_W + j];
    for (int m = threadIdx.x; m < M_; m += blockDim.x)
        wi[m] = word_i[g * M_ + m];
    __syncthreads();

    float acc = 0.0f;
    for (int idx = threadIdx.x; idx < KDIM * M_; idx += blockDim.x) {
        const int k = idx >> 6;       // idx / 64
        const int m = idx & (M_ - 1); // idx % 64
        const float d = yj[k] - W[(size_t)k * N_W + wi[m]];
        acc += d * d;
    }
    const float S = block_reduce(acc, lds);
    __syncthreads();

    float swv = 0.0f;
    if (threadIdx.x < M_) swv = Sw[(size_t)j * N_W + wi[threadIdx.x]];
    const float swsum = block_reduce(swv, lds);

    if (threadIdx.x == 0) gterm[g] = sqrtf(S) * swsum;
}

// ---------------- diff_e: single tiny workgroup -----------------------------

__global__ void diffe_kernel(const float* __restrict__ E,
                             const float* __restrict__ Se,
                             const int* __restrict__ ej,
                             const int* __restrict__ row_ind_p,
                             float* __restrict__ eterm) {
    const int r = row_ind_p[0];

    __shared__ float erow[KDIM];
    __shared__ float lds[8];

    for (int k = threadIdx.x; k < KDIM; k += blockDim.x)
        erow[k] = E[(size_t)r * KDIM + k];
    __syncthreads();

    float acc = 0.0f;
    for (int idx = threadIdx.x; idx < J_ * KDIM; idx += blockDim.x) {
        const int jj = idx >> 7;          // idx / 128
        const int k  = idx & (KDIM - 1);  // idx % 128
        const float d = erow[k] - E[(size_t)ej[jj] * KDIM + k];
        acc += d * d;
    }
    const float S = block_reduce(acc, lds);
    __syncthreads();

    float sev = (threadIdx.x < J_) ? Se[(size_t)r * N_E + ej[threadIdx.x]] : 0.0f;
    const float ses = block_reduce(sev, lds);

    if (threadIdx.x == 0) eterm[0] = sqrtf(S) * ses;
}

// ---------------- finalize: fixed-order combine (deterministic) -------------

__global__ void finalize_kernel(const float* __restrict__ reconP, int nr,
                                const float* __restrict__ wP, int nwp,
                                const float* __restrict__ eP, int nep,
                                const float* __restrict__ gterm,
                                const float* __restrict__ eterm,
                                const float* __restrict__ lamb,
                                float* __restrict__ out) {
    __shared__ float lds[8];
    const int tid = threadIdx.x;
    const int bd  = blockDim.x;
    float v;

    v = 0.0f; for (int i = tid; i < nr; i += bd) v += reconP[i];
    const float recon_ss = block_reduce(v, lds);
    __syncthreads();

    v = 0.0f; for (int i = tid; i < nwp; i += bd) v += wP[i];
    const float w_ss = block_reduce(v, lds);
    __syncthreads();

    v = 0.0f; for (int i = tid; i < nep; i += bd) v += eP[i];
    const float e_ss = block_reduce(v, lds);
    __syncthreads();

    v = 0.0f; for (int i = tid; i < G_; i += bd) v += gterm[i];
    const float simw = block_reduce(v, lds);

    if (tid == 0) {
        out[0] = sqrtf(recon_ss) +
                 lamb[0] * (sqrtf(w_ss) + sqrtf(e_ss)) +
                 simw + eterm[0];
    }
}

// ---------------- launcher ---------------------------------------------------

extern "C" void kernel_launch(void* const* d_in, const int* in_sizes, int n_in,
                              void* d_out, int out_size, void* d_ws, size_t ws_size,
                              hipStream_t stream) {
    const float* actual = (const float*)d_in[0];
    const float* pred   = (const float*)d_in[1];
    const float* W      = (const float*)d_in[2];
    const float* E      = (const float*)d_in[3];
    const float* Sw     = (const float*)d_in[4];
    const float* Se     = (const float*)d_in[5];
    const float* lamb   = (const float*)d_in[6];
    const int*   row    = (const int*)d_in[7];
    const int*   word_i = (const int*)d_in[8];
    const int*   ent_j  = (const int*)d_in[9];
    const int*   samp_j = (const int*)d_in[10];

    float* out = (float*)d_out;
    float* ws  = (float*)d_ws;

    // Workspace layout (floats); every slot written every call -> no init needed.
    const int NB_R = 1024;  // recon partials
    const int NB_W = 64;    // relu(W) partials
    const int NB_E = 32;    // relu(E) partials
    float* reconP = ws;
    float* wP     = reconP + NB_R;
    float* eP     = wP + NB_W;
    float* gterm  = eP + NB_E;          // G_ entries
    float* eterm  = gterm + G_;         // 1 entry

    const int ntiles_recon = (int)(((long long)N_E * N_W) / 512);  // 65536
    const int ntiles_W     = (KDIM * N_W) / 512;                   // 2048
    const int ntiles_E     = (N_E * KDIM) / 512;                   // 1024

    sumsq_diff_wmma<<<NB_R, 256, 0, stream>>>(actual, pred, ntiles_recon, reconP);
    sumsq_relu_wmma<<<NB_W, 256, 0, stream>>>(W, ntiles_W, wP);
    sumsq_relu_wmma<<<NB_E, 256, 0, stream>>>(E, ntiles_E, eP);
    diffw_kernel<<<G_, 256, 0, stream>>>(W, Sw, word_i, samp_j, gterm);
    diffe_kernel<<<1, 256, 0, stream>>>(E, Se, ent_j, row, eterm);
    finalize_kernel<<<1, 256, 0, stream>>>(reconP, NB_R, wP, NB_W, eP, NB_E,
                                           gterm, eterm, lamb, out);
}
